// TransformerSquared_75247827026071
// MI455X (gfx1250) — matrix operand order repaired and checked by reading, not compiled, not run on hardware
//
#include <hip/hip_runtime.h>
#include <math.h>

typedef __attribute__((ext_vector_type(16))) _Float16 v16h;
typedef __attribute__((ext_vector_type(8)))  _Float16 v8h;
typedef __attribute__((ext_vector_type(8)))  float    v8f;

#define DIMD   1024
#define NHEAD  16
#define DH     64
#define BATCH  4
#define SEQ    1024
#define ROWS   (BATCH * SEQ)   /* 4096 */
#define SCALE  0.125f          /* 1/sqrt(64) */
#define LN_EPS 1e-5f

// ---------------------------------------------------------------------------
// CDNA5 async global->LDS DMA (ASYNCcnt-tracked), per ISA 08_async_tensor §4.
// lds_dst: generic pointer to __shared__ (low 32 bits == LDS byte address).
// ---------------------------------------------------------------------------
__device__ __forceinline__ void async_copy_b128(void* lds_dst, const void* gsrc) {
  asm volatile("global_load_async_to_lds_b128 %0, %1, off"
               :
               : "v"((uint32_t)(uintptr_t)lds_dst),
                 "v"((uint64_t)(uintptr_t)gsrc)
               : "memory");
}

__device__ __forceinline__ void wait_async0() {
#if __has_builtin(__builtin_amdgcn_s_wait_asynccnt)
  __builtin_amdgcn_s_wait_asynccnt(0);
#else
  asm volatile("s_wait_asynccnt 0x0" ::: "memory");
#endif
}

// ---------------------------------------------------------------------------
// WMMA helpers (CDNA5 gfx1250: D = A(16x32,f16) * B(32x16,f16) + C(16x16,f32))
// ---------------------------------------------------------------------------
__device__ __forceinline__ v8f wmma_f32_16x16x32(v16h a, v16h b, v8f c) {
  return __builtin_amdgcn_wmma_f32_16x16x32_f16(
      /*neg_a=*/false, a, /*neg_b=*/false, b,
      /*c_mod=*/(short)0, c, /*reuse_a=*/false, /*reuse_b=*/false);
}

// A fragment: source tile is row-major [M][K] with leading dim ldk (halves).
// ISA layout: lane L<16 holds row M=L, K = {0..7, 16..23}; lane L+16 holds
// row M=L, K = {8..15, 24..31}.  -> two 16B contiguous LDS loads per lane.
__device__ __forceinline__ v16h load_frag_a(const _Float16* base, int ldk) {
  const int lane = threadIdx.x & 31;
  const int r = lane & 15, hh = lane >> 4;
  const _Float16* p = base + r * ldk + hh * 8;
  v8h lo = *(const v8h*)(p);        // K = hh*8 .. hh*8+7
  v8h hi = *(const v8h*)(p + 16);   // K = 16 + hh*8 ..
  return __builtin_shufflevector(lo, hi, 0, 1, 2, 3, 4, 5, 6, 7,
                                 8, 9, 10, 11, 12, 13, 14, 15);
}

// B fragment: source tile is [N][K] row-major with leading dim ldk (halves).
// ISA layout: lane L<16 -> col N=L, K={0..7,16..23}; lane L+16 -> col N=L, K={8..15,24..31}.
__device__ __forceinline__ v16h load_frag_b(const _Float16* base, int ldk) {
  const int lane = threadIdx.x & 31;
  const int n = lane & 15, hh = lane >> 4;
  const _Float16* p = base + n * ldk + hh * 8;
  v8h lo = *(const v8h*)(p);
  v8h hi = *(const v8h*)(p + 16);
  return __builtin_shufflevector(lo, hi, 0, 1, 2, 3, 4, 5, 6, 7,
                                 8, 9, 10, 11, 12, 13, 14, 15);
}

// ---------------------------------------------------------------------------
// Kernel 0: svd_adapt[d] = sigma * sum_r U[d,r] * (sum_j V[r,j])
// ---------------------------------------------------------------------------
__global__ __launch_bounds__(256)
void adapt_kernel(const float* __restrict__ sigma, const float* __restrict__ U,
                  const float* __restrict__ V, float* __restrict__ adapt) {
  __shared__ float part[256];
  __shared__ float rs[4];
  const int tid = threadIdx.x;
  const int r = tid >> 6;      // rank 0..3, 64 threads each
  const int c0 = tid & 63;
  float sum = 0.f;
  for (int c = c0; c < DIMD; c += 64) sum += V[r * DIMD + c];
  part[tid] = sum;
  __syncthreads();
  for (int o = 32; o > 0; o >>= 1) {
    if ((tid & 63) < o) part[tid] += part[tid + o];
    __syncthreads();
  }
  if ((tid & 63) == 0) rs[r] = part[tid];
  __syncthreads();
  const float sg = sigma[0];
  for (int d = tid; d < DIMD; d += 256) {
    float a = 0.f;
#pragma unroll
    for (int k = 0; k < 4; ++k) a += U[d * 4 + k] * rs[k];
    adapt[d] = sg * a;
  }
}

// ---------------------------------------------------------------------------
// Kernel 1: f32 -> f16 weight conversion
// ---------------------------------------------------------------------------
__global__ void f2h_kernel(const float* __restrict__ src,
                           _Float16* __restrict__ dst, int n) {
  int i = blockIdx.x * blockDim.x + threadIdx.x;
  const int stride = gridDim.x * blockDim.x;
  for (; i < n; i += stride) dst[i] = (_Float16)src[i];
}

// ---------------------------------------------------------------------------
// Generic WMMA GEMM:  C[M=4096][N] = A[M][K] * B^T  (+ bias)
//   A: f32 (adapt broadcast add + f16 convert in A-stage) or f16 (async DMA)
//   B: f16, stored [N][K] row-major; staged with global_load_async_to_lds_b128
//   C: f16 or f32
// Block tile 128(M) x 64(N); 8 waves, each 32x32 (4 accumulators); K-step 32.
// Double-buffered LDS: DMA of tile i+1 overlaps WMMA on tile i.
// ---------------------------------------------------------------------------
template <bool A_F32, bool C_F32>
__global__ __launch_bounds__(256)
void gemm_wmma(const void* __restrict__ Av, int lda, int acol,
               const float* __restrict__ adapt,
               const _Float16* __restrict__ Bw, int ldb,
               const float* __restrict__ bias,
               void* __restrict__ Cv, int ldc, int K) {
  __shared__ __align__(16) _Float16 As[2][128][32];
  __shared__ __align__(16) _Float16 Bs[2][64][32];

  const int tid = threadIdx.x;
  const int lane = tid & 31;
  const int wid = tid >> 5;
  const int wm = wid >> 1;  // 0..3 (M)
  const int wn = wid & 1;   // 0..1 (N)
  const int blockM = blockIdx.y * 128;
  const int blockN = blockIdx.x * 64;

  const float* Af = (const float*)Av;
  const _Float16* Ah = (const _Float16*)Av;

  const int arow = tid >> 1;          // 0..127
  const int acseg = (tid & 1) * 16;   // 0 or 16
  const int brow = tid >> 2;          // 0..63
  const int bcseg = (tid & 3) * 8;    // 0,8,16,24

  v8f acc[2][2] = {};

  auto stage = [&](int kt, int buf) {
    if (A_F32) {
      const float* srcA = Af + (size_t)(blockM + arow) * lda + acol + kt + acseg;
      float4 f0 = *(const float4*)(srcA + 0);
      float4 f1 = *(const float4*)(srcA + 4);
      float4 f2 = *(const float4*)(srcA + 8);
      float4 f3 = *(const float4*)(srcA + 12);
      const float* ad = adapt + acol + kt + acseg;
      v8h lo, hi;
      lo[0] = (_Float16)(f0.x + ad[0]);
      lo[1] = (_Float16)(f0.y + ad[1]);
      lo[2] = (_Float16)(f0.z + ad[2]);
      lo[3] = (_Float16)(f0.w + ad[3]);
      lo[4] = (_Float16)(f1.x + ad[4]);
      lo[5] = (_Float16)(f1.y + ad[5]);
      lo[6] = (_Float16)(f1.z + ad[6]);
      lo[7] = (_Float16)(f1.w + ad[7]);
      hi[0] = (_Float16)(f2.x + ad[8]);
      hi[1] = (_Float16)(f2.y + ad[9]);
      hi[2] = (_Float16)(f2.z + ad[10]);
      hi[3] = (_Float16)(f2.w + ad[11]);
      hi[4] = (_Float16)(f3.x + ad[12]);
      hi[5] = (_Float16)(f3.y + ad[13]);
      hi[6] = (_Float16)(f3.z + ad[14]);
      hi[7] = (_Float16)(f3.w + ad[15]);
      *(v8h*)&As[buf][arow][acseg] = lo;
      *(v8h*)&As[buf][arow][acseg + 8] = hi;
    } else {
      const _Float16* srcA = Ah + (size_t)(blockM + arow) * lda + acol + kt + acseg;
      async_copy_b128(&As[buf][arow][acseg], srcA);
      async_copy_b128(&As[buf][arow][acseg + 8], srcA + 8);
    }
    const _Float16* srcB = Bw + (size_t)(blockN + brow) * ldb + kt + bcseg;
    async_copy_b128(&Bs[buf][brow][bcseg], srcB);
  };

  const int nk = K >> 5;
  stage(0, 0);
  wait_async0();
  __syncthreads();

  for (int i = 0; i < nk; ++i) {
    if (i + 1 < nk) stage((i + 1) << 5, (i + 1) & 1);
    const int b = i & 1;
    v16h a0 = load_frag_a(&As[b][wm * 32][0], 32);
    v16h a1 = load_frag_a(&As[b][wm * 32 + 16][0], 32);
    v16h b0 = load_frag_b(&Bs[b][wn * 32][0], 32);
    v16h b1 = load_frag_b(&Bs[b][wn * 32 + 16][0], 32);
    acc[0][0] = wmma_f32_16x16x32(a0, b0, acc[0][0]);
    acc[0][1] = wmma_f32_16x16x32(a0, b1, acc[0][1]);
    acc[1][0] = wmma_f32_16x16x32(a1, b0, acc[1][0]);
    acc[1][1] = wmma_f32_16x16x32(a1, b1, acc[1][1]);
    wait_async0();
    __syncthreads();
  }

  // ---- epilogue: C layout: VGPR j -> row = j + 8*(lane>=16), col = lane&15
  const int hh = lane >> 4, nn = lane & 15;
#pragma unroll
  for (int ti = 0; ti < 2; ++ti)
#pragma unroll
    for (int tj = 0; tj < 2; ++tj) {
      const int col = blockN + wn * 32 + tj * 16 + nn;
      const float bv = bias ? bias[col] : 0.0f;
#pragma unroll
      for (int j = 0; j < 8; ++j) {
        const int row = blockM + wm * 32 + ti * 16 + hh * 8 + j;
        const float v = acc[ti][tj][j] + bv;
        if (C_F32)
          ((float*)Cv)[(size_t)row * ldc + col] = v;
        else
          ((_Float16*)Cv)[(size_t)row * ldc + col] = (_Float16)v;
      }
    }
}

// ---------------------------------------------------------------------------
// Differential attention: one block per (batch, head, 32-query tile).
// P = softmax(Q1K1^T/8) - softmax(Q2K2^T/8);  O = P @ (0.5*(V1+V2))
// Dynamic LDS: sc[32][1024]h + Pd[32][1024]h + Qt[2][32][64]h + Kt[64][64]h
//            = 144 KB (gfx1250 WGP has 320 KB).
// Q/K tiles staged with async global->LDS DMA.
// ---------------------------------------------------------------------------
__global__ __launch_bounds__(256)
void attn_diff(const _Float16* __restrict__ QKV1,
               const _Float16* __restrict__ QKV2,
               _Float16* __restrict__ O) {
  extern __shared__ __align__(16) char smem_raw[];
  _Float16* sc = (_Float16*)smem_raw;   // [32][1024] raw scores (half)
  _Float16* Pd = sc + 32 * 1024;        // [32][1024] P1 then P1-P2
  _Float16* Qt = Pd + 32 * 1024;        // [2][32][64] Q tiles (both streams)
  _Float16* Kt = Qt + 2 * 32 * 64;      // [64][64] K tile / [64][32] W^T tile

  const int tid = threadIdx.x;
  const int lane = tid & 31;
  const int wid = tid >> 5;
  const int mi = wid & 1;   // 0..1 : 16-row tile
  const int ni = wid >> 1;  // 0..3 : 16-col tile
  const int hh = lane >> 4, nn = lane & 15;

  const int qbase = blockIdx.x * 32;
  const int head = blockIdx.y;
  const size_t rowbase = (size_t)blockIdx.z * SEQ;
  const int qcol = head * DH;
  const int kcol = DIMD + head * DH;
  const int vcol = 2 * DIMD + head * DH;

  // async-load Q tiles for both streams: 2 * 32 * 64 halves
  {
    const int s = tid >> 7;        // stream 0/1 (128 threads each)
    const int idx = tid & 127;
    const int r = idx >> 2;        // 0..31
    const int c = (idx & 3) * 16;  // 0,16,32,48
    const _Float16* src =
        (s ? QKV2 : QKV1) + (rowbase + qbase + r) * 3072 + qcol + c;
    async_copy_b128(&Qt[(s * 32 + r) * 64 + c], src);
    async_copy_b128(&Qt[(s * 32 + r) * 64 + c + 8], src + 8);
  }
  wait_async0();
  __syncthreads();

  for (int s = 0; s < 2; ++s) {
    const _Float16* QKV = s ? QKV2 : QKV1;
    // ---- scores: S[32][1024] = Q_s @ K_s^T * SCALE ----
    for (int kt = 0; kt < SEQ; kt += 64) {
      {
        const int r = tid >> 2;        // key 0..63
        const int c = (tid & 3) * 16;  // dh chunk
        const _Float16* src = QKV + (rowbase + kt + r) * 3072 + kcol + c;
        async_copy_b128(&Kt[r * 64 + c], src);
        async_copy_b128(&Kt[r * 64 + c + 8], src + 8);
      }
      wait_async0();
      __syncthreads();
      v8f acc = {};
      // B matrix = K^T: Kt is [keyN][dhK] == the [N][K] layout load_frag_b wants
      acc = wmma_f32_16x16x32(load_frag_a(Qt + (s * 32 + mi * 16) * 64, 64),
                              load_frag_b(Kt + (ni * 16) * 64, 64), acc);
      acc = wmma_f32_16x16x32(load_frag_a(Qt + (s * 32 + mi * 16) * 64 + 32, 64),
                              load_frag_b(Kt + (ni * 16) * 64 + 32, 64), acc);
#pragma unroll
      for (int j = 0; j < 8; ++j)
        sc[(mi * 16 + hh * 8 + j) * 1024 + kt + ni * 16 + nn] =
            (_Float16)(acc[j] * SCALE);
      __syncthreads();
    }
    // ---- row-wise softmax (8 threads per row, shfl_xor over lanes 1/2/4) ----
    {
      const int r = tid >> 3, c0 = tid & 7;
      float m = -1e30f;
      for (int c = c0; c < SEQ; c += 8) m = fmaxf(m, (float)sc[r * 1024 + c]);
      m = fmaxf(m, __shfl_xor(m, 1));
      m = fmaxf(m, __shfl_xor(m, 2));
      m = fmaxf(m, __shfl_xor(m, 4));
      float z = 0.f;
      for (int c = c0; c < SEQ; c += 8) z += __expf((float)sc[r * 1024 + c] - m);
      z += __shfl_xor(z, 1);
      z += __shfl_xor(z, 2);
      z += __shfl_xor(z, 4);
      const float inv = 1.0f / z;
      for (int c = c0; c < SEQ; c += 8) {
        const float p = __expf((float)sc[r * 1024 + c] - m) * inv;
        Pd[r * 1024 + c] = s ? (_Float16)((float)Pd[r * 1024 + c] - p)
                             : (_Float16)p;
      }
    }
    __syncthreads();
  }

  // ---- O[32][64] = Pd[32][1024] @ W[1024][64], W = 0.5*(V1+V2) ----
  v8f acc = {};
  _Float16* Wt = Kt;  // reuse as W^T tile [dh=64][key=32]
  for (int kt = 0; kt < SEQ; kt += 32) {
    __syncthreads();
    {
      const int n = tid & 63;         // dh
      const int k0 = (tid >> 6) * 8;  // key sub-chunk
#pragma unroll
      for (int j = 0; j < 8; ++j) {
        const size_t rr = (rowbase + kt + k0 + j) * 3072 + vcol + n;
        Wt[n * 32 + k0 + j] =
            (_Float16)(0.5f * ((float)QKV1[rr] + (float)QKV2[rr]));
      }
    }
    __syncthreads();
    acc = wmma_f32_16x16x32(load_frag_a(Pd + (mi * 16) * 1024 + kt, 1024),
                            load_frag_b(Wt + (ni * 16) * 32, 32), acc);
  }
#pragma unroll
  for (int j = 0; j < 8; ++j) {
    const int row = qbase + mi * 16 + hh * 8 + j;
    const int col = head * DH + ni * 16 + nn;
    O[(rowbase + row) * (size_t)DIMD + col] = (_Float16)acc[j];
  }
}

// ---------------------------------------------------------------------------
// LayerNorm over rows of 1024 (bias already added by out-proj GEMM).
// ---------------------------------------------------------------------------
__global__ __launch_bounds__(256)
void ln_kernel(const float* __restrict__ Y, const float* __restrict__ g,
               const float* __restrict__ b, float* __restrict__ out) {
  __shared__ float red[8];
  const int row = blockIdx.x;
  const int tid = threadIdx.x;
  const int lane = tid & 31, wid = tid >> 5;
  const float* y = Y + (size_t)row * DIMD;
  float vals[4];
  float s = 0.f;
#pragma unroll
  for (int j = 0; j < 4; ++j) { vals[j] = y[tid + j * 256]; s += vals[j]; }
#pragma unroll
  for (int o = 16; o > 0; o >>= 1) s += __shfl_xor(s, o);
  if (lane == 0) red[wid] = s;
  __syncthreads();
  float tot = 0.f;
#pragma unroll
  for (int w = 0; w < 8; ++w) tot += red[w];
  const float mu = tot * (1.0f / 1024.0f);
  __syncthreads();
  float s2 = 0.f;
#pragma unroll
  for (int j = 0; j < 4; ++j) { const float d = vals[j] - mu; s2 += d * d; }
#pragma unroll
  for (int o = 16; o > 0; o >>= 1) s2 += __shfl_xor(s2, o);
  if (lane == 0) red[wid] = s2;
  __syncthreads();
  float tot2 = 0.f;
#pragma unroll
  for (int w = 0; w < 8; ++w) tot2 += red[w];
  const float rstd = rsqrtf(tot2 * (1.0f / 1024.0f) + LN_EPS);
#pragma unroll
  for (int j = 0; j < 4; ++j) {
    const int c = tid + j * 256;
    out[(size_t)row * DIMD + c] = (vals[j] - mu) * rstd * g[c] + b[c];
  }
}

// ---------------------------------------------------------------------------
// Host launch
// ---------------------------------------------------------------------------
extern "C" void kernel_launch(void* const* d_in, const int* in_sizes, int n_in,
                              void* d_out, int out_size, void* d_ws,
                              size_t ws_size, hipStream_t stream) {
  (void)in_sizes; (void)n_in; (void)out_size; (void)ws_size;
  const float* x   = (const float*)d_in[0];
  const float* sig = (const float*)d_in[1];
  const float* U   = (const float*)d_in[2];
  const float* V   = (const float*)d_in[3];
  const float* Wp1 = (const float*)d_in[4];
  const float* bp1 = (const float*)d_in[5];
  const float* Wp2 = (const float*)d_in[6];
  const float* bp2 = (const float*)d_in[7];
  const float* Wq1 = (const float*)d_in[8];
  const float* Wq2 = (const float*)d_in[9];
  const float* Wo  = (const float*)d_in[10];
  const float* bo  = (const float*)d_in[11];
  const float* lng = (const float*)d_in[12];
  const float* lnb = (const float*)d_in[13];
  float* out = (float*)d_out;

  char* ws = (char*)d_ws;
  size_t off = 0;
  auto carve = [&](size_t bytes) -> char* {
    char* p = ws + off;
    off += (bytes + 255) & ~(size_t)255;
    return p;
  };
  float*    adapt = (float*)carve(DIMD * 4);
  _Float16* Wp1h = (_Float16*)carve((size_t)1024 * 512 * 2);
  _Float16* Wp2h = (_Float16*)carve((size_t)1024 * 512 * 2);
  _Float16* Wq1h = (_Float16*)carve((size_t)3072 * 1024 * 2);
  _Float16* Wq2h = (_Float16*)carve((size_t)3072 * 1024 * 2);
  _Float16* Woh  = (_Float16*)carve((size_t)1024 * 1024 * 2);
  _Float16* X1p  = (_Float16*)carve((size_t)ROWS * 1024 * 2);
  _Float16* X2p  = (_Float16*)carve((size_t)ROWS * 1024 * 2);
  _Float16* QKV1 = (_Float16*)carve((size_t)ROWS * 3072 * 2);
  _Float16* QKV2 = (_Float16*)carve((size_t)ROWS * 3072 * 2);
  _Float16* Obuf = (_Float16*)carve((size_t)ROWS * 1024 * 2);
  float*    OUTF = (float*)carve((size_t)ROWS * 1024 * 4);

  dim3 blk(256);

  adapt_kernel<<<1, blk, 0, stream>>>(sig, U, V, adapt);
  f2h_kernel<<<256, blk, 0, stream>>>(Wp1, Wp1h, 1024 * 512);
  f2h_kernel<<<256, blk, 0, stream>>>(Wp2, Wp2h, 1024 * 512);
  f2h_kernel<<<256, blk, 0, stream>>>(Wq1, Wq1h, 3072 * 1024);
  f2h_kernel<<<256, blk, 0, stream>>>(Wq2, Wq2h, 3072 * 1024);
  f2h_kernel<<<256, blk, 0, stream>>>(Wo, Woh, 1024 * 1024);

  // x1p = (x[:, :512] + adapt[:512]) @ Wp1^T + bp1 ; x2p likewise
  gemm_wmma<true, false><<<dim3(16, 32), blk, 0, stream>>>(
      x, 1024, 0, adapt, Wp1h, 512, bp1, X1p, 1024, 512);
  gemm_wmma<true, false><<<dim3(16, 32), blk, 0, stream>>>(
      x, 1024, 512, adapt, Wp2h, 512, bp2, X2p, 1024, 512);
  // qkv_i = x_ip @ Wqkv_i^T  -> [4096][3072] (q|k|v)
  gemm_wmma<false, false><<<dim3(48, 32), blk, 0, stream>>>(
      X1p, 1024, 0, nullptr, Wq1h, 1024, nullptr, QKV1, 3072, 1024);
  gemm_wmma<false, false><<<dim3(48, 32), blk, 0, stream>>>(
      X2p, 1024, 0, nullptr, Wq2h, 1024, nullptr, QKV2, 3072, 1024);

  const size_t attn_smem =
      (size_t)(32 * 1024 + 32 * 1024 + 2 * 32 * 64 + 64 * 64) * sizeof(_Float16);
  (void)hipFuncSetAttribute((const void*)attn_diff,
                            hipFuncAttributeMaxDynamicSharedMemorySize,
                            (int)attn_smem);
  attn_diff<<<dim3(SEQ / 32, NHEAD, BATCH), blk, attn_smem, stream>>>(QKV1, QKV2,
                                                                      Obuf);

  // out = O @ Wo^T + bo  (f32 accumulate/out), then LN
  gemm_wmma<false, true><<<dim3(16, 32), blk, 0, stream>>>(
      Obuf, 1024, 0, nullptr, Woh, 1024, bo, OUTF, 1024, 1024);
  ln_kernel<<<ROWS, blk, 0, stream>>>(OUTF, lng, lnb, out);
}
